// BiLSTM_CRF_80444737454099
// MI455X (gfx1250) — compile-verified, matrix operand order checked by reading, not run on hardware
//
#include <hip/hip_runtime.h>
#include <hip/hip_bf16.h>

typedef _Float16 v16h __attribute__((ext_vector_type(16)));
typedef float    v8f  __attribute__((ext_vector_type(8)));

#define Vv 32000
#define Ee 300
#define EP 320          // E padded to multiple of 32
#define HH 256          // hidden per direction
#define G4 1024         // 4*HH
#define Bb 32
#define Ss 512
#define Tt 12
#define START_TAG 10
#define STOP_TAG 11
#define NEGV -10000.0f

__device__ __forceinline__ v8f wmma16x16x32(v16h a, v16h b, v8f c) {
    return __builtin_amdgcn_wmma_f32_16x16x32_f16(false, a, false, b, (short)0, c, false, false);
}
__device__ __forceinline__ float sigm(float x) { return 1.0f / (1.0f + __expf(-x)); }

// load 16 halves in WMMA operand order: elements 0..7 = [base..base+7],
// elements 8..15 = [base+16..base+23]  (two 16B vector loads)
template <typename PTR>
__device__ __forceinline__ v16h load_operand(PTR p) {
    v16h r;
#pragma unroll
    for (int i = 0; i < 8; ++i) r[i] = p[i];
#pragma unroll
    for (int i = 0; i < 8; ++i) r[8 + i] = p[16 + i];
    return r;
}

// ---------------------------------------------------------------------------
// Kernel 1: convert weights to f16 (wih padded 300->320 with zeros, whh 256)
// ---------------------------------------------------------------------------
__global__ void convert_weights(const float* wih_f, const float* whh_f,
                                const float* wih_b, const float* whh_b,
                                _Float16* wihf_f, _Float16* whhf_f,
                                _Float16* wihf_b, _Float16* whhf_b) {
    int id = blockIdx.x * blockDim.x + threadIdx.x;
    const int nWih = G4 * EP;          // 327680
    const int nWhh = G4 * HH;          // 262144
    if (id < nWih) {
        int row = id / EP, k = id % EP;
        float vf = (k < Ee) ? wih_f[row * Ee + k] : 0.0f;
        float vb = (k < Ee) ? wih_b[row * Ee + k] : 0.0f;
        wihf_f[id] = (_Float16)vf;
        wihf_b[id] = (_Float16)vb;
    } else if (id < nWih + nWhh) {
        int j = id - nWih;
        whhf_f[j] = (_Float16)whh_f[j];
        whhf_b[j] = (_Float16)whh_b[j];
    }
}

// ---------------------------------------------------------------------------
// Kernel 1b: gather embeddings -> padded f16 activation matrix xf16[16384][320]
//   row = s*32 + b  (xt layout [S,B,E]); k padded 300 -> 320 with zeros.
//   Coalesced, branch-free staging so the GEMM A-operand is pure b128 loads.
// ---------------------------------------------------------------------------
__global__ void gather_embed(const int* __restrict__ inputs,
                             const float* __restrict__ word_embed,
                             _Float16* __restrict__ xf16) {
    int id = blockIdx.x * blockDim.x + threadIdx.x;   // [0, 16384*320)
    if (id >= Ss * Bb * EP) return;
    const int k = id % EP;
    const int row = id / EP;
    const int s = row >> 5, b = row & 31;
    const int token = inputs[b * Ss + s];
    const float v = (k < Ee) ? word_embed[(size_t)token * Ee + k] : 0.0f;
    xf16[id] = (_Float16)v;
}

// ---------------------------------------------------------------------------
// Kernel 2: igates = xf16 @ wih.T + bih + bhh   (one launch / direction)
//   rows = [S*B=16384], cols = 1024 gates, K = 320
//   8 waves/block, each wave owns (mtile, 4 consecutive ntiles)
// ---------------------------------------------------------------------------
__global__ __launch_bounds__(256) void igates_gemm(
    const _Float16* __restrict__ xf16,
    const _Float16* __restrict__ wihf, const float* __restrict__ bih,
    const float* __restrict__ bhh, float* __restrict__ igates) {
    const int wave = threadIdx.x >> 5;
    const int lane = threadIdx.x & 31;
    const int task = blockIdx.x * 8 + wave;   // 16384 tasks
    const int mtile = task >> 4;              // [0,1024)
    const int ng    = task & 15;              // group of 4 ntiles
    const int l16   = lane & 15;
    const int khalf = (lane & 16) ? 8 : 0;
    const int rsel  = (lane & 16) ? 8 : 0;

    const _Float16* arow = xf16 + (size_t)(mtile * 16 + l16) * EP;

    v8f acc[4] = {v8f{0}, v8f{0}, v8f{0}, v8f{0}};

    for (int kt = 0; kt < 10; ++kt) {
        const int base = kt * 32 + khalf;
        const v16h a = load_operand(arow + base);
#pragma unroll
        for (int g = 0; g < 4; ++g) {
            const int n = (ng * 4 + g) * 16 + l16;
            const v16h bv = load_operand(wihf + (size_t)n * EP + base);
            acc[g] = wmma16x16x32(a, bv, acc[g]);
        }
    }
#pragma unroll
    for (int g = 0; g < 4; ++g) {
        const int n = (ng * 4 + g) * 16 + l16;
        const float bias = bih[n] + bhh[n];
#pragma unroll
        for (int r = 0; r < 8; ++r) {
            const int m = mtile * 16 + r + rsel;
            igates[(size_t)m * G4 + n] = acc[g][r] + bias;
        }
    }
}

// ---------------------------------------------------------------------------
// Kernel 3: persistent recurrent LSTM, grid=2 (blockIdx.x = direction)
//   per step: gates = h@whh.T + igates[s]; LSTM cell; h,c in LDS (48 KB)
//   wave (mtile, jt) computes i/f/g/o tiles for its own hidden columns,
//   so the cell update is done entirely in registers (no gate staging).
// ---------------------------------------------------------------------------
__global__ __launch_bounds__(1024) void lstm_recurrent(
    const _Float16* __restrict__ whhf_f, const float* __restrict__ ig_f,
    float* __restrict__ hs_f,
    const _Float16* __restrict__ whhf_b, const float* __restrict__ ig_b,
    float* __restrict__ hs_b) {
    const int dir = blockIdx.x;
    const _Float16* whhf = dir ? whhf_b : whhf_f;
    const float* igates  = dir ? ig_b   : ig_f;
    float* hs            = dir ? hs_b   : hs_f;

    __shared__ _Float16 hf16[Bb * HH];   // 16 KB
    __shared__ float    csh[Bb * HH];    // 32 KB

    for (int i = threadIdx.x; i < Bb * HH; i += 1024) {
        hf16[i] = (_Float16)0.0f;
        csh[i] = 0.0f;
    }
    __syncthreads();

    const int wave = threadIdx.x >> 5;
    const int lane = threadIdx.x & 31;
    const int mtile = wave & 1;          // batch tile (0..1)
    const int jt    = wave >> 1;         // hidden column tile (0..15)
    const int l16   = lane & 15;
    const int khalf = (lane & 16) ? 8 : 0;
    const int rsel  = (lane & 16) ? 8 : 0;

    for (int si = 0; si < Ss; ++si) {
        const int s = dir ? (Ss - 1 - si) : si;

        v8f acc[4] = {v8f{0}, v8f{0}, v8f{0}, v8f{0}};
        for (int kt = 0; kt < 8; ++kt) {
            const int kbase = kt * 32 + khalf;
            const v16h a = load_operand(&hf16[(mtile * 16 + l16) * HH + kbase]);
#pragma unroll
            for (int g = 0; g < 4; ++g) {
                const int n = g * HH + jt * 16 + l16;
                const v16h bv = load_operand(whhf + (size_t)n * HH + kbase);
                acc[g] = wmma16x16x32(a, bv, acc[g]);
            }
        }
        __syncthreads();   // all reads of hf16 (this step) complete

        const float* ig = igates + (size_t)s * Bb * G4;
        const int j = jt * 16 + l16;
#pragma unroll
        for (int r = 0; r < 8; ++r) {
            const int m = mtile * 16 + r + rsel;
            const size_t mb = (size_t)m * G4;
            const float gi = acc[0][r] + ig[mb + j];
            const float gf = acc[1][r] + ig[mb + HH + j];
            const float gg = acc[2][r] + ig[mb + 2 * HH + j];
            const float go = acc[3][r] + ig[mb + 3 * HH + j];
            const int hidx = m * HH + j;
            const float c = sigm(gf) * csh[hidx] + sigm(gi) * tanhf(gg);
            const float h = sigm(go) * tanhf(c);
            csh[hidx] = c;
            hf16[hidx] = (_Float16)h;
            hs[((size_t)s * Bb + m) * HH + j] = h;
        }
        __syncthreads();   // new h visible for next step
    }
}

// ---------------------------------------------------------------------------
// Kernel 4: feats = concat(hf,hb) @ fc_w.T + fc_b     (T=12, plain VALU)
// ---------------------------------------------------------------------------
__global__ void fc_kernel(const float* __restrict__ hs_f, const float* __restrict__ hs_b,
                          const float* __restrict__ fc_w, const float* __restrict__ fc_b,
                          float* __restrict__ feats) {
    int id = blockIdx.x * blockDim.x + threadIdx.x;
    if (id >= Bb * Ss * Tt) return;
    const int t = id % Tt;
    const int s = (id / Tt) % Ss;
    const int b = id / (Tt * Ss);
    const float* hfp = hs_f + ((size_t)s * Bb + b) * HH;
    const float* hbp = hs_b + ((size_t)s * Bb + b) * HH;
    const float* wf = fc_w + t * 512;
    float sum = fc_b[t];
    for (int k = 0; k < HH; ++k) sum += hfp[k] * wf[k];
    for (int k = 0; k < HH; ++k) sum += hbp[k] * wf[HH + k];
    feats[((size_t)b * Ss + s) * Tt + t] = sum;
}

// ---------------------------------------------------------------------------
// Kernel 5: Viterbi per sequence (32 blocks x 32 threads)
// ---------------------------------------------------------------------------
__global__ void viterbi_kernel(const float* __restrict__ feats,
                               const float* __restrict__ trans,
                               int* __restrict__ bps, float* __restrict__ out) {
    const int b = blockIdx.x;
    const int tid = threadIdx.x;
    __shared__ float tr[Tt * Tt];
    __shared__ float prev[Tt], nxt[Tt];
    for (int i = tid; i < Tt * Tt; i += 32) tr[i] = trans[i];
    if (tid < Tt) prev[tid] = (tid == START_TAG) ? 0.0f : NEGV;
    __syncthreads();

    for (int s = 0; s < Ss; ++s) {
        if (tid < Tt) {
            float best = prev[0] + tr[tid];
            int arg = 0;
            for (int p = 1; p < Tt; ++p) {
                float v = prev[p] + tr[p * Tt + tid];
                if (v > best) { best = v; arg = p; }
            }
            nxt[tid] = best + feats[((size_t)b * Ss + s) * Tt + tid];
            bps[((size_t)b * Ss + s) * Tt + tid] = arg;
        }
        __syncthreads();
        if (tid < Tt) prev[tid] = nxt[tid];
        __syncthreads();
    }

    if (tid == 0) {
        float best = prev[0] + tr[0 * Tt + STOP_TAG];
        int last = 0;
        for (int t = 1; t < Tt; ++t) {
            float v = prev[t] + tr[t * Tt + STOP_TAG];
            if (v > best) { best = v; last = t; }
        }
        out[b] = best;                                   // scores[b]
        float* path = out + Bb + (size_t)b * Ss;
        int tag = last;
        path[Ss - 1] = (float)tag;
        for (int s = Ss - 1; s >= 1; --s) {
            tag = bps[((size_t)b * Ss + s) * Tt + tag];
            path[s - 1] = (float)tag;
        }
    }
}

// ---------------------------------------------------------------------------
extern "C" void kernel_launch(void* const* d_in, const int* in_sizes, int n_in,
                              void* d_out, int out_size, void* d_ws, size_t ws_size,
                              hipStream_t stream) {
    const int*   inputs     = (const int*)  d_in[0];
    const float* word_embed = (const float*)d_in[1];
    const float* wih_f = (const float*)d_in[2];
    const float* whh_f = (const float*)d_in[3];
    const float* bih_f = (const float*)d_in[4];
    const float* bhh_f = (const float*)d_in[5];
    const float* wih_b = (const float*)d_in[6];
    const float* whh_b = (const float*)d_in[7];
    const float* bih_b = (const float*)d_in[8];
    const float* bhh_b = (const float*)d_in[9];
    const float* fc_w  = (const float*)d_in[10];
    const float* fc_b  = (const float*)d_in[11];
    const float* trans = (const float*)d_in[12];

    char* ws = (char*)d_ws;
    size_t off = 0;
    auto carve = [&](size_t bytes) { void* p = ws + off; off += (bytes + 255) & ~(size_t)255; return p; };
    _Float16* wihf16_f = (_Float16*)carve((size_t)G4 * EP * 2);
    _Float16* wihf16_b = (_Float16*)carve((size_t)G4 * EP * 2);
    _Float16* whhf16_f = (_Float16*)carve((size_t)G4 * HH * 2);
    _Float16* whhf16_b = (_Float16*)carve((size_t)G4 * HH * 2);
    _Float16* xf16     = (_Float16*)carve((size_t)Ss * Bb * EP * 2);  // 10.5 MB
    float* igates_f = (float*)carve((size_t)Ss * Bb * G4 * 4);   // 64 MB
    float* igates_b = (float*)carve((size_t)Ss * Bb * G4 * 4);   // 64 MB
    float* hs_f     = (float*)carve((size_t)Ss * Bb * HH * 4);   // 16 MB
    float* hs_b     = (float*)carve((size_t)Ss * Bb * HH * 4);   // 16 MB
    float* feats    = (float*)carve((size_t)Bb * Ss * Tt * 4);
    int*   bps      = (int*)  carve((size_t)Bb * Ss * Tt * 4);

    // 1) weight conversion to f16
    {
        int total = G4 * EP + G4 * HH;
        convert_weights<<<(total + 255) / 256, 256, 0, stream>>>(
            wih_f, whh_f, wih_b, whh_b, wihf16_f, whhf16_f, wihf16_b, whhf16_b);
    }
    // 1b) stage gathered embeddings as padded f16 (branch-free GEMM operands)
    {
        int total = Ss * Bb * EP;
        gather_embed<<<(total + 255) / 256, 256, 0, stream>>>(inputs, word_embed, xf16);
    }
    // 2) input projections (WMMA GEMM), one per direction
    igates_gemm<<<2048, 256, 0, stream>>>(xf16, wihf16_f, bih_f, bhh_f, igates_f);
    igates_gemm<<<2048, 256, 0, stream>>>(xf16, wihf16_b, bih_b, bhh_b, igates_b);
    // 3) both recurrent directions concurrently (grid = 2)
    lstm_recurrent<<<2, 1024, 0, stream>>>(whhf16_f, igates_f, hs_f,
                                           whhf16_b, igates_b, hs_b);
    // 4) FC -> CRF emission feats
    fc_kernel<<<(Bb * Ss * Tt + 255) / 256, 256, 0, stream>>>(hs_f, hs_b, fc_w, fc_b, feats);
    // 5) Viterbi decode (scores + paths into d_out)
    viterbi_kernel<<<Bb, 32, 0, stream>>>(feats, trans, bps, (float*)d_out);
}
